// BQuantConv1d_simple2_10273561772176
// MI455X (gfx1250) — matrix operand order repaired and checked by reading, not compiled, not run on hardware
//
#include <hip/hip_runtime.h>
#include <hip/hip_bf16.h>

// Problem constants (from reference)
#define NF    4096
#define BITS  8
#define NX    4096
#define BATCH 4
#define SEQ   2048
#define M_DIM (BATCH * SEQ)   // 8192
#define N_DIM NF              // 4096
#define K_DIM NX              // 4096

typedef __bf16 v16bf __attribute__((ext_vector_type(16)));
typedef __bf16 v8bf  __attribute__((ext_vector_type(8)));
typedef float  v8f   __attribute__((ext_vector_type(8)));

union BF16Frag { v16bf v; v8bf h[2]; };

// ---- CDNA5 async global->LDS path (ASYNCcnt-tracked), guarded ----
#if __has_builtin(__builtin_amdgcn_global_load_async_to_lds_b128) && \
    __has_builtin(__builtin_amdgcn_s_wait_asynccnt)
#define HAVE_ASYNC_LDS 1
typedef int v4i32 __attribute__((vector_size(16)));
typedef __attribute__((address_space(1))) v4i32* as1_v4i;  // global int4*
typedef __attribute__((address_space(3))) v4i32* as3_v4i;  // LDS int4*
#else
#define HAVE_ASYNC_LDS 0
#endif

static __device__ __forceinline__ void async_copy16(const void* g, void* l) {
#if HAVE_ASYNC_LDS
  __builtin_amdgcn_global_load_async_to_lds_b128(
      (as1_v4i)(void*)g, (as3_v4i)l, /*imm offset*/0, /*cpol*/0);
#else
  *(uint4*)l = *(const uint4*)g;
#endif
}

static __device__ __forceinline__ void wait_async_done() {
#if HAVE_ASYNC_LDS
  __builtin_amdgcn_s_wait_asynccnt(0);
#endif
}

static __device__ __forceinline__ unsigned short f32_to_bf16(float f) {
  // round-to-nearest-even truncation
  unsigned int u = __float_as_uint(f);
  unsigned int r = (u + 0x7FFFu + ((u >> 16) & 1u)) >> 16;
  return (unsigned short)r;
}

// ---------------------------------------------------------------------------
// Kernel 1: dequantize packed sign bits -> bf16 weight matrix W[N_DIM][K_DIM]
// ---------------------------------------------------------------------------
__global__ __launch_bounds__(256) void bq_dequant_kernel(
    const float* __restrict__ scale,          // [NF, 1, BITS]
    const unsigned char* __restrict__ binary, // [NF, BITS, NX/8]
    unsigned short* __restrict__ wbf)         // [NF, NX] bf16 bits
{
  int t = blockIdx.x * 256 + threadIdx.x;     // 0 .. NF*(NX/8)
  int f = t >> 9;                              // NX/8 == 512
  int p = t & 511;

  float        s[BITS];
  unsigned int byt[BITS];
#pragma unroll
  for (int b = 0; b < BITS; ++b) {
    s[b]   = scale[f * BITS + b];
    byt[b] = binary[((size_t)f * BITS + b) * (NX / 8) + p];
  }

  unsigned short w16[8];
#pragma unroll
  for (int j = 0; j < 8; ++j) {
    unsigned int bit = 1u << (7 - j);          // mask order 128..1
    float acc = 0.0f;
#pragma unroll
    for (int b = 0; b < BITS; ++b)
      acc += (byt[b] & bit) ? s[b] : -s[b];
    w16[j] = f32_to_bf16(acc);
  }

  uint4 o;
  o.x = (unsigned)w16[0] | ((unsigned)w16[1] << 16);
  o.y = (unsigned)w16[2] | ((unsigned)w16[3] << 16);
  o.z = (unsigned)w16[4] | ((unsigned)w16[5] << 16);
  o.w = (unsigned)w16[6] | ((unsigned)w16[7] << 16);
  *(uint4*)&wbf[(size_t)f * K_DIM + p * 8] = o;
}

// ---------------------------------------------------------------------------
// Kernel 2: convert activations f32 -> bf16
// ---------------------------------------------------------------------------
__global__ __launch_bounds__(256) void bq_cvt_x_kernel(
    const float* __restrict__ x, unsigned short* __restrict__ xbf)
{
  size_t t = (size_t)blockIdx.x * 256 + threadIdx.x;   // one thread per 8 elems
  const float4* xp = (const float4*)x;
  float4 a = xp[t * 2 + 0];
  float4 b = xp[t * 2 + 1];
  uint4 o;
  o.x = (unsigned)f32_to_bf16(a.x) | ((unsigned)f32_to_bf16(a.y) << 16);
  o.y = (unsigned)f32_to_bf16(a.z) | ((unsigned)f32_to_bf16(a.w) << 16);
  o.z = (unsigned)f32_to_bf16(b.x) | ((unsigned)f32_to_bf16(b.y) << 16);
  o.w = (unsigned)f32_to_bf16(b.z) | ((unsigned)f32_to_bf16(b.w) << 16);
  *(uint4*)&xbf[t * 8] = o;
}

// ---------------------------------------------------------------------------
// Kernel 3: bf16 WMMA GEMM: out[M,N] = X[M,K] * W[N,K]^T + bias
// 256 threads = 8 wave32s (2x4 grid), block tile 128(M) x 256(N),
// wave tile 64x64 = 4x4 v_wmma_f32_16x16x32_bf16.
// Double-buffered LDS fed by async global->LDS copies; one barrier per K-tile.
// ---------------------------------------------------------------------------
#define BK  64
#define LDT 72   // padded row stride (bf16 elems): 144 B, keeps 16B alignment

__global__ __launch_bounds__(256) void bq_gemm_kernel(
    const unsigned short* __restrict__ xbf,   // [M,K] bf16
    const unsigned short* __restrict__ wbf,   // [N,K] bf16
    const float* __restrict__ bias,           // [N]
    float* __restrict__ out)                  // [M,N] f32
{
  __shared__ __align__(16) unsigned short As[2][128 * LDT];
  __shared__ __align__(16) unsigned short Bs[2][256 * LDT];

  const int tid  = threadIdx.x;
  const int lane = tid & 31;
  const int wave = tid >> 5;
  const int lm   = lane & 15;   // row/col within 16x16 tile
  const int lh   = lane >> 4;   // lane half selects K sub-range / M sub-rows

  const int m0 = blockIdx.y * 128;
  const int n0 = blockIdx.x * 256;
  const int wm = (wave >> 2) * 64;   // wave origin in block tile (M)
  const int wn = (wave & 3)  * 64;   // wave origin in block tile (N)

  // staging coordinates: chunk c = tid + i*256 -> row = rbase + i*32, col kcA
  const int rbase = tid >> 3;
  const int kcA   = (tid & 7) * 8;

  v8f acc[4][4] = {};

  // issue per-thread async copies of one K-tile into buffer `buf`
  auto stage = [&](int buf, int kt) {
    const unsigned short* ga = &xbf[(size_t)(m0 + rbase) * K_DIM + kt + kcA];
    unsigned short*       la = &As[buf][rbase * LDT + kcA];
#pragma unroll
    for (int i = 0; i < 4; ++i)   // A: 128 rows x 64 K (16 KB)
      async_copy16(ga + (size_t)i * 32 * K_DIM, la + i * 32 * LDT);
    const unsigned short* gb = &wbf[(size_t)(n0 + rbase) * K_DIM + kt + kcA];
    unsigned short*       lb = &Bs[buf][rbase * LDT + kcA];
#pragma unroll
    for (int i = 0; i < 8; ++i)   // B: 256 rows x 64 K (32 KB)
      async_copy16(gb + (size_t)i * 32 * K_DIM, lb + i * 32 * LDT);
  };

  stage(0, 0);                    // prologue: fill buffer 0
  int cur = 0;

  for (int kt = 0; kt < K_DIM; kt += BK) {
    // my async loads for buf[cur] done; barrier publishes LDS to all waves and
    // guarantees every wave finished reading buf[cur^1] (results already in VGPRs)
    wait_async_done();
    __syncthreads();

    if (kt + BK < K_DIM)
      stage(cur ^ 1, kt + BK);    // overlap next tile's loads with compute

    // speculative cache prefetch two tiles ahead (global_prefetch_b8)
    if (kt + 2 * BK < K_DIM) {
      __builtin_prefetch(&xbf[(size_t)(m0 + (tid >> 1)) * K_DIM + kt + 2 * BK + (tid & 1) * 32], 0, 1);
      __builtin_prefetch(&wbf[(size_t)(n0 + tid) * K_DIM + kt + 2 * BK], 0, 1);
    }

#pragma unroll
    for (int ks = 0; ks < BK; ks += 32) {
      BF16Frag a[4], b[4];
      // A 16x32 bf16 fragment: lanes 0-15 -> K ks+{0..7, 16..23},
      //                        lanes 16-31 -> K ks+{8..15, 24..31}
#pragma unroll
      for (int mi = 0; mi < 4; ++mi) {
        const unsigned short* base = &As[cur][(wm + mi * 16 + lm) * LDT + ks + lh * 8];
        a[mi].h[0] = *(const v8bf*)(base);
        a[mi].h[1] = *(const v8bf*)(base + 16);
      }
      // B 32x16 bf16 fragment: lane = column (row of W); lanes 0-15 K ks+{0..7, 16..23},
      //                        lanes 16-31 K ks+{8..15, 24..31}
#pragma unroll
      for (int ni = 0; ni < 4; ++ni) {
        const unsigned short* base = &Bs[cur][(wn + ni * 16 + lm) * LDT + ks + lh * 8];
        b[ni].h[0] = *(const v8bf*)(base);
        b[ni].h[1] = *(const v8bf*)(base + 16);
      }
#pragma unroll
      for (int mi = 0; mi < 4; ++mi)
#pragma unroll
        for (int ni = 0; ni < 4; ++ni)
          acc[mi][ni] = __builtin_amdgcn_wmma_f32_16x16x32_bf16(
              false, a[mi].v, false, b[ni].v,
              (short)0, acc[mi][ni], false, false);
    }
    cur ^= 1;
  }

  // ---- epilogue: bias add + store. C/D layout: lane = N column (lm),
  // VGPR v = M row v (lanes 0-15) or v+8 (lanes 16-31).
#pragma unroll
  for (int ni = 0; ni < 4; ++ni) {
    int n = n0 + wn + ni * 16 + lm;
    float bv = bias[n];
#pragma unroll
    for (int mi = 0; mi < 4; ++mi) {
      int mbase = m0 + wm + mi * 16 + lh * 8;
#pragma unroll
      for (int v = 0; v < 8; ++v) {
        out[(size_t)(mbase + v) * N_DIM + n] = acc[mi][ni][v] + bv;
      }
    }
  }
}

// ---------------------------------------------------------------------------
extern "C" void kernel_launch(void* const* d_in, const int* in_sizes, int n_in,
                              void* d_out, int out_size, void* d_ws, size_t ws_size,
                              hipStream_t stream) {
  const float*         x      = (const float*)d_in[0];         // [4,2048,4096] f32
  const float*         scale  = (const float*)d_in[1];         // [4096,1,8] f32
  const float*         bias   = (const float*)d_in[2];         // [4096] f32
  const unsigned char* binary = (const unsigned char*)d_in[3]; // [4096,8,512,1] i8
  float*               out    = (float*)d_out;                 // [4,2048,4096] f32

  // workspace: W bf16 [N,K] then X bf16 [M,K]
  unsigned short* wbf = (unsigned short*)d_ws;
  unsigned short* xbf = (unsigned short*)((char*)d_ws + (size_t)N_DIM * K_DIM * 2);

  // 1) dequantize weights: NF * (NX/8) threads
  bq_dequant_kernel<<<(NF * (NX / 8)) / 256, 256, 0, stream>>>(scale, binary, wbf);
  // 2) convert x to bf16: (M*K)/8 threads
  bq_cvt_x_kernel<<<((size_t)M_DIM * K_DIM / 8) / 256, 256, 0, stream>>>(x, xbf);
  // 3) GEMM with bias
  bq_gemm_kernel<<<dim3(N_DIM / 256, M_DIM / 128), 256, 0, stream>>>(xbf, wbf, bias, out);
}
